// SAModuleRatio_80272938762721
// MI455X (gfx1250) — matrix operand order repaired and checked by reading, not compiled, not run on hardware
//
#include <hip/hip_runtime.h>

// ---------------- problem constants (match reference) ----------------
#define BC    4        // clouds
#define NBP   4096     // points per cloud
#define MBP   1024     // centroids per cloud
#define NG    (BC*MBP) // 4096 total centroids
#define NPT   (BC*NBP) // 16384 total points
#define KNB   64       // max neighbors per ball
#define DIN   64       // input feature dim
#define KP    96       // padded layer-1 K dim (67 -> 96 = 3 * 32)
#define APADS 104      // Apad LDS row stride (halves): 52 DW -> conflict-free
#define HS    136      // Hb1/Hb2 LDS row stride (halves): 68 DW -> conflict-free
#define H1    128
#define H2    128
#define H3    256
#define RAD2  0.04f    // R^2 = 0.2^2

typedef __attribute__((ext_vector_type(16))) _Float16 v16h;
typedef __attribute__((ext_vector_type(8)))  float    v8f;

union HV { unsigned u[8]; v16h h; };

// ---- A fragment (16x32 f16) from LDS, row-major, 'stride' halves/row ----
// ISA 7.12.2: lane L -> M = L%16; VGPR v -> K = 2v (v<4) or 16+2(v-4) (v>=4),
// +8 for lanes 16..31. Each VGPR holds halves K, K+1 (one b32 LDS read).
__device__ __forceinline__ v16h load_a_lds(const _Float16* base, int stride, int lane) {
  const int m  = lane & 15;
  const int hi = (lane >> 4) & 1;
  const _Float16* row = base + m * stride;
  HV r;
#pragma unroll
  for (int v = 0; v < 8; ++v) {
    int k = ((v < 4) ? (2 * v) : (16 + 2 * (v - 4))) + (hi ? 8 : 0);
    r.u[v] = *(const unsigned*)(row + k);
  }
  return r.h;
}

// ---- B fragment (32x16 f16) from transposed weights Wt[N][Kdim] ----
// lane L -> N = L%16; VGPR v -> K = 2v (v<4) or 16+2(v-4) (v>=4),
// +8 for lanes 16..31 (same K order as the A fragment).
__device__ __forceinline__ v16h load_b_any(const _Float16* wt, int kdim,
                                           int nbase, int kbase, int lane) {
  const int n  = lane & 15;
  const int kh = (lane >> 4) ? 8 : 0;
  const _Float16* row = wt + (size_t)(nbase + n) * kdim + kbase + kh;
  HV r;
#pragma unroll
  for (int v = 0; v < 8; ++v) r.u[v] = *(const unsigned*)(row + ((v < 4) ? (2 * v) : (16 + 2 * (v - 4))));
  return r.h;
}

__device__ __forceinline__ v8f wmma_f16(v16h a, v16h b, v8f c) {
  return __builtin_amdgcn_wmma_f32_16x16x32_f16(false, a, false, b,
                                                (short)0, c, false, false);
}

// ---- gfx1250 async global->LDS copy (VGLOBAL op 98) + counter fence ----
// Per ISA 10.2 aperture rule, LDS accesses use the generic address's low 32 bits, so the low 32
// bits of a generic pointer to __shared__ are the wave-relative LDS address.
__device__ __forceinline__ void async_g2l_b128(unsigned lds_addr, const void* gptr) {
  unsigned long long ga = (unsigned long long)(uintptr_t)gptr;
  asm volatile("global_load_async_to_lds_b128 %0, %1, off"
               :: "v"(lds_addr), "v"(ga) : "memory");
}
__device__ __forceinline__ void wait_asynccnt0() {
  asm volatile("s_wait_asynccnt 0" ::: "memory");
}

// ------- kernel 0: weight transpose/pad/convert + x -> f16 convert ----------
__global__ __launch_bounds__(256) void prep_kernel(
    const float* __restrict__ x,
    const float* __restrict__ W1, const float* __restrict__ W2,
    const float* __restrict__ W3,
    unsigned short* __restrict__ Wt1u, unsigned short* __restrict__ Wt2u,
    unsigned short* __restrict__ Wt3u, unsigned short* __restrict__ xhu) {
  int idx = blockIdx.x * 256 + threadIdx.x;
  _Float16* Wt1 = (_Float16*)Wt1u;
  _Float16* Wt2 = (_Float16*)Wt2u;
  _Float16* Wt3 = (_Float16*)Wt3u;
  _Float16* xh  = (_Float16*)xhu;
  const int S1 = H1 * KP;          // 128 x 96
  const int S2 = H2 * H1;          // 128 x 128
  const int S3 = H3 * H2;          // 256 x 128
  if (idx < S1) {
    int n = idx / KP, k = idx % KP;
    float v = (k < DIN + 3) ? W1[k * H1 + n] : 0.f;
    Wt1[idx] = (_Float16)v;
  } else if (idx < S1 + S2) {
    int i = idx - S1; int n = i / H1, k = i % H1;
    Wt2[i] = (_Float16)W2[k * H2 + n];
  } else if (idx < S1 + S2 + S3) {
    int i = idx - S1 - S2; int n = i / H2, k = i % H2;
    Wt3[i] = (_Float16)W3[k * H3 + n];
  } else if (idx < S1 + S2 + S3 + NPT * DIN) {
    int i = idx - S1 - S2 - S3;
    xh[i] = (_Float16)x[i];
  }
}

// ---------------- kernel 1: farthest point sampling (1 block / cloud) --------
// wave32 shuffle argmax: 5 xor steps in-wave, 1 LDS exchange, 3 xor steps in
// wave 0 -> only 2 barriers per sequential FPS step.
__global__ __launch_bounds__(256) void fps_kernel(
    const float* __restrict__ pos, int* __restrict__ sel,
    float* __restrict__ pos_out, int* __restrict__ batch_out) {
  __shared__ float px[NBP], py[NBP], pz[NBP];   // 48 KB
  __shared__ float wrv[8];
  __shared__ int   wri[8];
  __shared__ int   sels[MBP];                   // 4 KB
  __shared__ int   cur;

  const int cloud = blockIdx.x;
  const int base  = cloud * NBP;
  const int tid   = threadIdx.x;
  const int wave  = tid >> 5;
  const int lane  = tid & 31;

  for (int i = tid; i < NBP; i += 256) {
    px[i] = pos[(size_t)(base + i) * 3 + 0];
    py[i] = pos[(size_t)(base + i) * 3 + 1];
    pz[i] = pos[(size_t)(base + i) * 3 + 2];
  }
  float dist[16];
#pragma unroll
  for (int j = 0; j < 16; ++j) dist[j] = 3.4e38f;
  if (tid == 0) { cur = 0; sels[0] = 0; }
  __syncthreads();

  for (int k = 1; k < MBP; ++k) {
    const float lx = px[cur], ly = py[cur], lz = pz[cur];  // LDS broadcast
    float best = -1.f; int bidx = tid * 16;
    const int i0 = tid * 16;                               // contiguous block
#pragma unroll
    for (int j = 0; j < 16; ++j) {
      int i = i0 + j;
      float dx = px[i] - lx, dy = py[i] - ly, dz = pz[i] - lz;
      float nd = fminf(dist[j], dx * dx + dy * dy + dz * dz);
      dist[j] = nd;
      if (nd > best) { best = nd; bidx = i; }   // first max within block
    }
    // in-wave argmax (lower lane owns lower index -> tie-break idx<)
#pragma unroll
    for (int off = 16; off > 0; off >>= 1) {
      float ov = __shfl_xor(best, off, 32);
      int   oi = __shfl_xor(bidx, off, 32);
      if (ov > best || (ov == best && oi < bidx)) { best = ov; bidx = oi; }
    }
    if (lane == 0) { wrv[wave] = best; wri[wave] = bidx; }
    __syncthreads();
    if (wave == 0) {
      float v = (lane < 8) ? wrv[lane] : -1.f;
      int   i = (lane < 8) ? wri[lane] : 0;
#pragma unroll
      for (int off = 4; off > 0; off >>= 1) {
        float ov = __shfl_xor(v, off, 32);
        int   oi = __shfl_xor(i, off, 32);
        if (ov > v || (ov == v && oi < i)) { v = ov; i = oi; }
      }
      if (lane == 0) { cur = i; sels[k] = i; }
    }
    __syncthreads();
  }

  // write sel (global point index), pos_out, batch_out
  for (int m = tid; m < MBP; m += 256) {
    int sl = sels[m];
    int g  = cloud * MBP + m;
    sel[g] = base + sl;
    pos_out[(size_t)g * 3 + 0] = px[sl];
    pos_out[(size_t)g * 3 + 1] = py[sl];
    pos_out[(size_t)g * 3 + 2] = pz[sl];
    batch_out[g] = cloud;
  }
}

// ---------------- kernel 2: ordered radius search (1 wave / centroid) --------
__global__ __launch_bounds__(256) void radius_kernel(
    const float* __restrict__ pos, const int* __restrict__ sel,
    int* __restrict__ nbr, int* __restrict__ cnt_out) {
  const int g    = blockIdx.x * 8 + (threadIdx.x >> 5);
  const int lane = threadIdx.x & 31;
  const int s    = sel[g];
  const float qx = pos[(size_t)s * 3 + 0];
  const float qy = pos[(size_t)s * 3 + 1];
  const float qz = pos[(size_t)s * 3 + 2];
  const int base = (g >> 10) * NBP;   // cloud * NBP

  int cnt = 0;
  for (int c0 = 0; c0 < NBP && cnt < KNB; c0 += 32) {
    int i = base + c0 + lane;
    float dx = pos[(size_t)i * 3 + 0] - qx;
    float dy = pos[(size_t)i * 3 + 1] - qy;
    float dz = pos[(size_t)i * 3 + 2] - qz;
    bool inb = (dx * dx + dy * dy + dz * dz) <= RAD2;
    unsigned mask = (unsigned)__ballot(inb);
    int pre  = __popc(mask & ((1u << lane) - 1u));
    int slot = cnt + pre;
    if (inb && slot < KNB) nbr[(size_t)g * KNB + slot] = i;
    cnt += __popc(mask);
  }
  if (cnt > KNB) cnt = KNB;
  for (int t = cnt + lane; t < KNB; t += 32)  // pad with centroid itself
    nbr[(size_t)g * KNB + t] = s;
  if (lane == 0) cnt_out[g] = cnt;
}

// ---------------- kernel 3: WMMA MLP + masked max-pool (1 block / centroid) --
// LDS (~34 KB): poolA (Apad stride 104, later Hb2 stride 136) + Hb1 stride 136.
// Padded strides shift each row by 4 LDS banks -> conflict-free ds_load_b128
// A-fragment feeds. The 64 neighbor x-rows are gathered by the async engine
// (global_load_async_to_lds_b128, per-lane addresses) straight into Apad,
// overlapped with VALU writing the rel-position columns.
// Each wave owns one N-tile per layer (two in layer 3) and drives FOUR independent accumulator
// chains (one per M-tile): B fragment loaded once per kt, then 4 back-to-back
// independent v_wmma -> no f16 WMMA->WMMA RAW hazard NOPs (ISA 7.12.1).
__global__ __launch_bounds__(256) void conv_kernel(
    const unsigned short* __restrict__ xhu, const float* __restrict__ pos,
    const int* __restrict__ sel, const int* __restrict__ nbr,
    const int* __restrict__ cnt_in,
    const unsigned short* __restrict__ Wt1u,
    const unsigned short* __restrict__ Wt2u,
    const unsigned short* __restrict__ Wt3u,
    const float* __restrict__ b1, const float* __restrict__ b2,
    const float* __restrict__ b3, float* __restrict__ out_x) {
  __shared__ __align__(16) unsigned char poolA[KNB * HS * 2]; // 17408 B
  __shared__ __align__(16) _Float16 Hb1[KNB][HS];             // 17408 B

  _Float16 (*Apad)[APADS] = (_Float16 (*)[APADS])poolA; // live: gather..layer 1
  _Float16 (*Hb2)[HS]     = (_Float16 (*)[HS])poolA;    // live: layer 2..layer 3

  const _Float16* Wt1 = (const _Float16*)Wt1u;
  const _Float16* Wt2 = (const _Float16*)Wt2u;
  const _Float16* Wt3 = (const _Float16*)Wt3u;

  const int g    = blockIdx.x;
  const int tid  = threadIdx.x;
  const int wave = tid >> 5;
  const int lane = tid & 31;
  const int s    = sel[g];
  const int cnt  = cnt_in[g];
  const float qx = pos[(size_t)s * 3 + 0];
  const float qy = pos[(size_t)s * 3 + 1];
  const float qz = pos[(size_t)s * 3 + 2];

  // async gather: 64 rows x 128 B of f16 x-features -> Apad[r][0..63]
  // 512 b128 ops, 2 per thread; Apad row stride 208 B is 16-B aligned.
  {
    const unsigned abase = (unsigned)(uintptr_t)&Apad[0][0];
    const unsigned char* xsrc = (const unsigned char*)xhu;
#pragma unroll
    for (int t = 0; t < 2; ++t) {
      const int o    = tid + 256 * t;
      const int r    = o >> 3;
      const int part = o & 7;
      const int j    = nbr[(size_t)g * KNB + r];
      async_g2l_b128(abase + (unsigned)(r * (APADS * 2) + part * 16),
                     xsrc + (size_t)j * (DIN * 2) + part * 16);
    }
  }
  // prefetch all converted weights (24+32+64 KB = 960 lines)
  for (int ln = tid; ln < (H1 * KP + H2 * H1 + H3 * H2) * 2 / 128; ln += 256)
    __builtin_prefetch(((const unsigned char*)Wt1u) + ln * 128, 0, 3);

  // rel-position columns + zero padding: cols 64..95 (96..103 never read)
  {
    const int r  = tid >> 2;
    const int c0 = 64 + (tid & 3) * 8;
    const int j  = nbr[(size_t)g * KNB + r];
#pragma unroll
    for (int c = c0; c < c0 + 8; ++c) {
      float v = 0.f;
      if (c == 64)      v = pos[(size_t)j * 3 + 0] - qx;
      else if (c == 65) v = pos[(size_t)j * 3 + 1] - qy;
      else if (c == 66) v = pos[(size_t)j * 3 + 2] - qz;
      Apad[r][c] = (_Float16)v;
    }
  }
  wait_asynccnt0();   // this wave's async LDS gathers landed
  __syncthreads();    // everyone's landed + rel columns complete

  // ---- layer 1: (64x96) @ (96x128); wave -> nt, 4 acc chains over mtiles ----
  {
    const int nt = wave;
    v8f acc[4] = {};
#pragma unroll
    for (int kt = 0; kt < 3; ++kt) {
      const v16h bf = load_b_any(Wt1, KP, nt * 16, kt * 32, lane);
#pragma unroll
      for (int mt = 0; mt < 4; ++mt)
        acc[mt] = wmma_f16(load_a_lds(&Apad[mt * 16][kt * 32], APADS, lane), bf, acc[mt]);
    }
    const int n  = nt * 16 + (lane & 15);
    const float bb = b1[n];
#pragma unroll
    for (int mt = 0; mt < 4; ++mt) {
      const int mr = mt * 16 + ((lane >> 4) ? 8 : 0);  // C layout: VGPR r -> M=r(+8)
#pragma unroll
      for (int r = 0; r < 8; ++r) {
        float v = acc[mt][r] + bb;
        Hb1[mr + r][n] = (_Float16)(v > 0.f ? v : 0.f);
      }
    }
  }
  __syncthreads();    // Apad dead from here; its LDS becomes Hb2

  // ---- layer 2: (64x128) @ (128x128); 4 ktiles ----
  {
    const int nt = wave;
    v8f acc[4] = {};
#pragma unroll
    for (int kt = 0; kt < 4; ++kt) {
      const v16h bf = load_b_any(Wt2, H1, nt * 16, kt * 32, lane);
#pragma unroll
      for (int mt = 0; mt < 4; ++mt)
        acc[mt] = wmma_f16(load_a_lds(&Hb1[mt * 16][kt * 32], HS, lane), bf, acc[mt]);
    }
    const int n  = nt * 16 + (lane & 15);
    const float bb = b2[n];
#pragma unroll
    for (int mt = 0; mt < 4; ++mt) {
      const int mr = mt * 16 + ((lane >> 4) ? 8 : 0);
#pragma unroll
      for (int r = 0; r < 8; ++r) {
        float v = acc[mt][r] + bb;
        Hb2[mr + r][n] = (_Float16)(v > 0.f ? v : 0.f);
      }
    }
  }
  __syncthreads();

  // ---- layer 3: (64x128) @ (128x256) fused with masked column max-pool ----
#pragma unroll
  for (int t = 0; t < 2; ++t) {
    const int nt = wave + t * 8;          // 8 waves x 2 = 16 ntiles
    v8f acc[4] = {};
#pragma unroll
    for (int kt = 0; kt < 4; ++kt) {
      const v16h bf = load_b_any(Wt3, H2, nt * 16, kt * 32, lane);
#pragma unroll
      for (int mt = 0; mt < 4; ++mt)
        acc[mt] = wmma_f16(load_a_lds(&Hb2[mt * 16][kt * 32], HS, lane), bf, acc[mt]);
    }
    const int n  = nt * 16 + (lane & 15);
    const float bb = b3[n];
    float mx = -1e9f;
#pragma unroll
    for (int mt = 0; mt < 4; ++mt) {
      const int kb = mt * 16 + ((lane >> 4) ? 8 : 0);  // neighbor slot of acc[r]
#pragma unroll
      for (int r = 0; r < 8; ++r) {
        float v = acc[mt][r] + bb;
        v = v > 0.f ? v : 0.f;
        if (kb + r >= cnt) v = -1e9f;   // invalid slots can't win the max
        mx = fmaxf(mx, v);
      }
    }
    // lanes l and l^16 hold the same N, different M halves -> combine
    mx = fmaxf(mx, __shfl_xor(mx, 16, 32));
    if (lane < 16) out_x[(size_t)g * H3 + nt * 16 + lane] = mx;
  }
}

// ---------------- host launcher ----------------
extern "C" void kernel_launch(void* const* d_in, const int* in_sizes, int n_in,
                              void* d_out, int out_size, void* d_ws, size_t ws_size,
                              hipStream_t stream) {
  const float* x   = (const float*)d_in[0];
  const float* pos = (const float*)d_in[1];
  // d_in[2] = batch (implied by layout; unused)
  const float* W1  = (const float*)d_in[3];
  const float* b1  = (const float*)d_in[4];
  const float* W2  = (const float*)d_in[5];
  const float* b2  = (const float*)d_in[6];
  const float* W3  = (const float*)d_in[7];
  const float* b3  = (const float*)d_in[8];

  float* out      = (float*)d_out;
  float* out_x    = out;                               // [4096, 256]
  float* out_pos  = out + (size_t)NG * H3;             // [4096, 3]
  int*   out_bat  = (int*)(out + (size_t)NG * H3 + (size_t)NG * 3); // [4096]

  // workspace layout
  int* sel = (int*)d_ws;                               // 4096
  int* cnt = sel + NG;                                 // 4096
  int* nbr = cnt + NG;                                 // 4096*64
  unsigned short* Wt1 = (unsigned short*)(nbr + (size_t)NG * KNB); // 128*96
  unsigned short* Wt2 = Wt1 + H1 * KP;                 // 128*128
  unsigned short* Wt3 = Wt2 + H2 * H1;                 // 256*128
  unsigned short* xh  = Wt3 + H3 * H2;                 // 16384*64 f16

  const int ptot = H1 * KP + H2 * H1 + H3 * H2 + NPT * DIN;
  prep_kernel<<<(ptot + 255) / 256, 256, 0, stream>>>(x, W1, W2, W3,
                                                      Wt1, Wt2, Wt3, xh);
  fps_kernel<<<BC, 256, 0, stream>>>(pos, sel, out_pos, out_bat);
  radius_kernel<<<NG / 8, 256, 0, stream>>>(pos, sel, nbr, cnt);
  conv_kernel<<<NG, 256, 0, stream>>>(xh, pos, sel, nbr, cnt,
                                      Wt1, Wt2, Wt3, b1, b2, b3, out_x);
}
